// CausalSelfAttention_81346680586459
// MI455X (gfx1250) — matrix operand order repaired and checked by reading, not compiled, not run on hardware
//
#include <hip/hip_runtime.h>
#include <hip/hip_bf16.h>

#define DEV __device__ __forceinline__

typedef __bf16 bf16;
typedef __attribute__((ext_vector_type(16))) __bf16 v16bf;
typedef __attribute__((ext_vector_type(8)))  float  v8f;
typedef __attribute__((ext_vector_type(4)))  int    v4i;

static constexpr int BZ = 2, T = 2048, D = 1024, H = 16, HD = 64;
static constexpr int M  = BZ * T;     // 4096 rows
static constexpr int N3 = 3 * D;      // 3072 qkv outputs

// ------------------------------------------------ async global->LDS (CDNA5)
#if defined(__has_builtin)
#if __has_builtin(__builtin_amdgcn_global_load_async_to_lds_b128)
#define USE_ASYNC 1
#endif
#endif

#define AS1 __attribute__((address_space(1)))
#define AS3 __attribute__((address_space(3)))

DEV void cp16(const bf16* g, bf16* l) {
#if defined(USE_ASYNC)
  __builtin_amdgcn_global_load_async_to_lds_b128(
      (AS1 v4i*)(v4i*)(void*)(bf16*)g, (AS3 v4i*)(v4i*)(void*)l, 0, 0);
#else
  *(uint4*)l = *(const uint4*)g;
#endif
}

DEV void cp_wait() {
#if defined(USE_ASYNC)
  asm volatile("s_wait_asynccnt 0x0" ::: "memory");
#endif
}

// ---------------------------------------------------------------- WMMA helpers

DEV v8f wmma_bf16(v16bf a, v16bf b, v8f c) {
  return __builtin_amdgcn_wmma_f32_16x16x32_bf16(false, a, false, b, (short)0, c,
                                                 false, false);
}

union FragU { uint4 u[2]; v16bf v; };

// A fragment: 16x32 bf16, tile row-major [16][ld]
DEV v16bf frag_a(const bf16* tile, int ld, int lane) {
  const int rhalf = lane >> 4, mrow = lane & 15;
  const bf16* row = tile + mrow * ld + rhalf * 8;
  FragU f;
  f.u[0] = *(const uint4*)(row);
  f.u[1] = *(const uint4*)(row + 16);
  return f.v;
}

// B fragment: 32x16 bf16, tile row-major [32][ld] (K-major, N contiguous)
DEV v16bf frag_b(const bf16* tile, int ld, int lane) {
  const int rhalf = lane >> 4, ncol = lane & 15;
  v16bf v;
#pragma unroll
  for (int s = 0; s < 16; ++s) {
    const int k = (s < 8) ? rhalf * 8 + s : 16 + rhalf * 8 + (s - 8);
    v[s] = tile[k * ld + ncol];
  }
  return v;
}

DEV float hmax16(float v) {
#pragma unroll
  for (int m = 1; m < 16; m <<= 1) v = fmaxf(v, __shfl_xor(v, m, 32));
  return v;
}
DEV float hsum16(float v) {
#pragma unroll
  for (int m = 1; m < 16; m <<= 1) v += __shfl_xor(v, m, 32);
  return v;
}

// ---------------------------------------------------------------- converts

__global__ __launch_bounds__(256)
void f32_to_bf16_kernel(const float* __restrict__ in, bf16* __restrict__ out, int n) {
  int i = blockIdx.x * 256 + threadIdx.x;
  if (i < n) out[i] = (bf16)in[i];
}

// in: f32 [N][Dd] row-major -> out: bf16 [Dd][N] (K-major for B fragments)
__global__ __launch_bounds__(256)
void cvt_transpose_kernel(const float* __restrict__ in, bf16* __restrict__ out,
                          int Nn, int Dd) {
  int idx = blockIdx.x * 256 + threadIdx.x;
  if (idx >= Nn * Dd) return;
  int k = idx / Nn, n = idx - k * Nn;
  out[idx] = (bf16)in[(size_t)n * Dd + k];
}

// ---------------------------------------------------------------- QKV GEMM
// C[M,N3] = xb @ w^T + b ;  wabT is [D][N3] (K-major).  q,v -> [B,H,T,hd],
// k -> [B,H,hd,T].  Double-buffered LDS, async-pipelined, 1 barrier / K-step.
__global__ __launch_bounds__(256)
void qkv_gemm_kernel(const bf16* __restrict__ xb, const bf16* __restrict__ wabT,
                     const float* __restrict__ b_attn,
                     bf16* __restrict__ qb, bf16* __restrict__ kTb,
                     bf16* __restrict__ vb) {
  __shared__ bf16 As[2][128 * 64];   // [m][k]
  __shared__ bf16 Bs[2][64 * 64];    // [k][n]
  const int tid = threadIdx.x, lane = tid & 31, wave = tid >> 5;
  const int wm = wave >> 1, wn = wave & 1;
  const int m0 = blockIdx.y * 128, n0 = blockIdx.x * 64;
  const int srow = tid >> 3, sc8 = (tid & 7) * 8;

  auto stage = [&](int buf, int k0) {
#pragma unroll
    for (int i = 0; i < 4; ++i) {
      int row = srow + i * 32;
      cp16(&xb[(size_t)(m0 + row) * D + k0 + sc8], &As[buf][row * 64 + sc8]);
    }
#pragma unroll
    for (int i = 0; i < 2; ++i) {
      int kk = srow + i * 32;
      cp16(&wabT[(size_t)(k0 + kk) * N3 + n0 + sc8], &Bs[buf][kk * 64 + sc8]);
    }
  };

  stage(0, 0);
  v8f acc[2][2] = {};
  const int NK = D / 64;
  for (int it = 0; it < NK; ++it) {
    cp_wait();
    __syncthreads();
    if (it + 1 < NK) stage((it + 1) & 1, (it + 1) * 64);   // overlap next panel
    if (it + 2 < NK)
      __builtin_prefetch(&xb[(size_t)(m0 + (tid >> 1)) * D + (it + 2) * 64], 0, 0);
    const bf16* Ab = As[it & 1];
    const bf16* Bb = Bs[it & 1];
#pragma unroll
    for (int ks = 0; ks < 2; ++ks) {
      v16bf a[2], b[2];
#pragma unroll
      for (int mi = 0; mi < 2; ++mi)
        a[mi] = frag_a(&Ab[(wm * 32 + mi * 16) * 64 + ks * 32], 64, lane);
#pragma unroll
      for (int ni = 0; ni < 2; ++ni)
        b[ni] = frag_b(&Bb[(ks * 32) * 64 + wn * 32 + ni * 16], 64, lane);
#pragma unroll
      for (int mi = 0; mi < 2; ++mi)
#pragma unroll
        for (int ni = 0; ni < 2; ++ni)
          acc[mi][ni] = wmma_bf16(a[mi], b[ni], acc[mi][ni]);
    }
  }
  // epilogue: section is uniform per block (64-wide N tile within one 1024 slab)
  const int rhalf = lane >> 4, ncol = lane & 15;
  const int sec = n0 >> 10;
#pragma unroll
  for (int mi = 0; mi < 2; ++mi)
#pragma unroll
    for (int ni = 0; ni < 2; ++ni) {
      const int gn = n0 + wn * 32 + ni * 16 + ncol;
      const float bias = b_attn[gn];
      const int didx = gn & 1023, h = didx >> 6, dd = didx & 63;
      const int gm0 = m0 + wm * 32 + mi * 16 + rhalf * 8;
      const int bi = gm0 >> 11, t0 = gm0 & (T - 1);
      if (sec == 1) {        // K: lane's 8 outputs contiguous in t -> b128 store
        union { uint4 u; bf16 e[8]; } pk;
#pragma unroll
        for (int v = 0; v < 8; ++v) pk.e[v] = (bf16)(acc[mi][ni][v] + bias);
        *(uint4*)&kTb[((size_t)((bi * H + h) * HD + dd) * T) + t0] = pk.u;
      } else {
        bf16* dst = (sec == 0) ? qb : vb;
        size_t base = ((size_t)((bi * H + h) * T + t0) << 6) + dd;
#pragma unroll
        for (int v = 0; v < 8; ++v)
          dst[base + ((size_t)v << 6)] = (bf16)(acc[mi][ni][v] + bias);
      }
    }
}

// ---------------------------------------------------------------- attention
// grid (T/64, B*H), block 128 = 4 waves; double-buffered K/V, async-pipelined.
__global__ __launch_bounds__(128)
void attn_kernel(const bf16* __restrict__ qb, const bf16* __restrict__ kTb,
                 const bf16* __restrict__ vb, bf16* __restrict__ ctx) {
  __shared__ bf16 Qs[64 * 64];         // [q][d]
  __shared__ bf16 Kt[2][64 * 64];      // [d][j]
  __shared__ bf16 Vs[2][64 * 64];      // [j][d]
  __shared__ bf16 Ps[4 * 16 * 64];     // per-wave P tile [q][j]
  const int tid = threadIdx.x, lane = tid & 31, wave = tid >> 5;
  const int bh = blockIdx.y, b = bh >> 4, h = bh & 15;
  const int q0 = blockIdx.x * 64;
  const bf16* qh = qb + (size_t)bh * T * HD;
  const bf16* kh = kTb + (size_t)bh * HD * T;   // [d][t]
  const bf16* vh = vb + (size_t)bh * T * HD;
  const float c = 0.125f * 1.44269504088896340736f;  // 1/sqrt(hd) * log2(e)
  const int rhalf = lane >> 4, ncol = lane & 15;
  const int srow = tid >> 3, sc8 = (tid & 7) * 8;

  auto stageKV = [&](int buf, int j0) {
#pragma unroll
    for (int i = 0; i < 4; ++i) {
      int r = srow + i * 16;
      cp16(&vh[(size_t)(j0 + r) * HD + sc8], &Vs[buf][r * 64 + sc8]);
      cp16(&kh[(size_t)r * T + j0 + sc8], &Kt[buf][r * 64 + sc8]);
    }
  };

#pragma unroll
  for (int i = 0; i < 4; ++i) {        // stage Q once
    int r = srow + i * 16;
    cp16(&qh[(size_t)(q0 + r) * HD + sc8], &Qs[r * 64 + sc8]);
  }
  stageKV(0, 0);

  v16bf aq[2];
  v8f acc[4] = {};
  float ms[8], ls[8];
#pragma unroll
  for (int v = 0; v < 8; ++v) { ms[v] = -1e30f; ls[v] = 0.f; }

  const int NB = T / 64;
  for (int jb = 0; jb < NB; ++jb) {
    cp_wait();
    __syncthreads();
    if (jb == 0) {                     // Q fragments: j-loop invariant
#pragma unroll
      for (int ks = 0; ks < 2; ++ks)
        aq[ks] = frag_a(&Qs[wave * 16 * 64 + ks * 32], 64, lane);
    }
    if (jb + 1 < NB) stageKV((jb + 1) & 1, (jb + 1) * 64);  // overlap next panel
    const bf16* KtB = Kt[jb & 1];
    const bf16* VsB = Vs[jb & 1];

    v8f s[4];
#pragma unroll
    for (int nc = 0; nc < 4; ++nc) {   // S = Q K^T (8 wmma)
      v8f sa = {};
#pragma unroll
      for (int ks = 0; ks < 2; ++ks)
        sa = wmma_bf16(aq[ks], frag_b(&KtB[ks * 32 * 64 + nc * 16], 64, lane), sa);
      s[nc] = sa;
    }
    // online softmax per q-row
#pragma unroll
    for (int v = 0; v < 8; ++v) {
      float rm = fmaxf(fmaxf(s[0][v], s[1][v]), fmaxf(s[2][v], s[3][v]));
      rm = hmax16(rm);
      float mn = fmaxf(ms[v], rm);
      float alpha = exp2f((ms[v] - mn) * c);
      ms[v] = mn;
      float rs = 0.f;
#pragma unroll
      for (int nc = 0; nc < 4; ++nc) {
        float p = exp2f((s[nc][v] - mn) * c);
        rs += p;
        Ps[(wave * 16 + rhalf * 8 + v) * 64 + nc * 16 + ncol] = (bf16)p;
      }
      rs = hsum16(rs);
      ls[v] = ls[v] * alpha + rs;
#pragma unroll
      for (int dc = 0; dc < 4; ++dc) acc[dc][v] *= alpha;
    }
    // O += P V (8 wmma, P fragments reused across d-chunks)
    v16bf ap[2];
#pragma unroll
    for (int ks = 0; ks < 2; ++ks)
      ap[ks] = frag_a(&Ps[wave * 16 * 64 + ks * 32], 64, lane);
#pragma unroll
    for (int dc = 0; dc < 4; ++dc) {
      v8f a8 = acc[dc];
#pragma unroll
      for (int ks = 0; ks < 2; ++ks)
        a8 = wmma_bf16(ap[ks], frag_b(&VsB[ks * 32 * 64 + dc * 16], 64, lane), a8);
      acc[dc] = a8;
    }
  }
  // normalize + write ctx [B,T,H*hd] bf16
#pragma unroll
  for (int v = 0; v < 8; ++v) {
    float inv = 1.f / ls[v];
    int t = q0 + wave * 16 + rhalf * 8 + v;
    size_t base = ((size_t)(b * T + t)) * D + h * HD;
#pragma unroll
    for (int dc = 0; dc < 4; ++dc)
      ctx[base + dc * 16 + ncol] = (bf16)(acc[dc][v] * inv);
  }
}

// ---------------------------------------------------------------- proj GEMM
// y[M,D] = ctx @ w_proj^T + b_proj (f32).  wpT is [D][D] K-major.
__global__ __launch_bounds__(256)
void proj_gemm_kernel(const bf16* __restrict__ cb, const bf16* __restrict__ wpT,
                      const float* __restrict__ b_proj, float* __restrict__ y) {
  __shared__ bf16 As[2][128 * 64];
  __shared__ bf16 Bs[2][64 * 64];
  const int tid = threadIdx.x, lane = tid & 31, wave = tid >> 5;
  const int wm = wave >> 1, wn = wave & 1;
  const int m0 = blockIdx.y * 128, n0 = blockIdx.x * 64;
  const int srow = tid >> 3, sc8 = (tid & 7) * 8;

  auto stage = [&](int buf, int k0) {
#pragma unroll
    for (int i = 0; i < 4; ++i) {
      int row = srow + i * 32;
      cp16(&cb[(size_t)(m0 + row) * D + k0 + sc8], &As[buf][row * 64 + sc8]);
    }
#pragma unroll
    for (int i = 0; i < 2; ++i) {
      int kk = srow + i * 32;
      cp16(&wpT[(size_t)(k0 + kk) * D + n0 + sc8], &Bs[buf][kk * 64 + sc8]);
    }
  };

  stage(0, 0);
  v8f acc[2][2] = {};
  const int NK = D / 64;
  for (int it = 0; it < NK; ++it) {
    cp_wait();
    __syncthreads();
    if (it + 1 < NK) stage((it + 1) & 1, (it + 1) * 64);
    const bf16* Ab = As[it & 1];
    const bf16* Bb = Bs[it & 1];
#pragma unroll
    for (int ks = 0; ks < 2; ++ks) {
      v16bf a[2], b[2];
#pragma unroll
      for (int mi = 0; mi < 2; ++mi)
        a[mi] = frag_a(&Ab[(wm * 32 + mi * 16) * 64 + ks * 32], 64, lane);
#pragma unroll
      for (int ni = 0; ni < 2; ++ni)
        b[ni] = frag_b(&Bb[(ks * 32) * 64 + wn * 32 + ni * 16], 64, lane);
#pragma unroll
      for (int mi = 0; mi < 2; ++mi)
#pragma unroll
        for (int ni = 0; ni < 2; ++ni)
          acc[mi][ni] = wmma_bf16(a[mi], b[ni], acc[mi][ni]);
    }
  }
  const int rhalf = lane >> 4, ncol = lane & 15;
#pragma unroll
  for (int mi = 0; mi < 2; ++mi)
#pragma unroll
    for (int ni = 0; ni < 2; ++ni) {
      int gn = n0 + wn * 32 + ni * 16 + ncol;
      float bias = b_proj[gn];
#pragma unroll
      for (int v = 0; v < 8; ++v) {
        int gm = m0 + wm * 32 + mi * 16 + rhalf * 8 + v;
        y[(size_t)gm * D + gn] = acc[mi][ni][v] + bias;
      }
    }
}

// ---------------------------------------------------------------- launcher

extern "C" void kernel_launch(void* const* d_in, const int* in_sizes, int n_in,
                              void* d_out, int out_size, void* d_ws, size_t ws_size,
                              hipStream_t stream) {
  (void)in_sizes; (void)n_in; (void)out_size; (void)ws_size;
  const float* x      = (const float*)d_in[0];
  const float* w_attn = (const float*)d_in[1];
  const float* b_attn = (const float*)d_in[2];
  const float* w_proj = (const float*)d_in[3];
  const float* b_proj = (const float*)d_in[4];
  float* y = (float*)d_out;

  char* ws = (char*)d_ws;
  size_t off = 0;
  bf16* xb   = (bf16*)(ws + off); off += (size_t)M * D * 2;     //  8 MB
  bf16* wabT = (bf16*)(ws + off); off += (size_t)N3 * D * 2;    //  6 MB  [D][N3]
  bf16* wpT  = (bf16*)(ws + off); off += (size_t)D * D * 2;     //  2 MB  [D][D]
  bf16* qb   = (bf16*)(ws + off); off += (size_t)M * D * 2;     //  8 MB  [B,H,T,hd]
  bf16* kTb  = (bf16*)(ws + off); off += (size_t)M * D * 2;     //  8 MB  [B,H,hd,T]
  bf16* vb   = (bf16*)(ws + off); off += (size_t)M * D * 2;     //  8 MB  [B,H,T,hd]
  bf16* ctx  = (bf16*)(ws + off); off += (size_t)M * D * 2;     //  8 MB  [B,T,D]

  f32_to_bf16_kernel<<<(M * D + 255) / 256, 256, 0, stream>>>(x, xb, M * D);
  cvt_transpose_kernel<<<(N3 * D + 255) / 256, 256, 0, stream>>>(w_attn, wabT, N3, D);
  cvt_transpose_kernel<<<(D * D + 255) / 256, 256, 0, stream>>>(w_proj, wpT, D, D);

  qkv_gemm_kernel<<<dim3(N3 / 64, M / 128), 256, 0, stream>>>(xb, wabT, b_attn,
                                                              qb, kTb, vb);
  attn_kernel<<<dim3(T / 64, BZ * H), 128, 0, stream>>>(qb, kTb, vb, ctx);
  proj_gemm_kernel<<<dim3(D / 64, M / 128), 256, 0, stream>>>(ctx, wpT, b_proj, y);
}
